// RNN_79525614452981
// MI455X (gfx1250) — matrix operand order repaired and checked by reading, not compiled, not run on hardware
//
#include <hip/hip_runtime.h>
#include <hip/hip_bf16.h>

typedef __attribute__((ext_vector_type(16))) _Float16 v16h;
typedef __attribute__((ext_vector_type(8)))  _Float16 v8h;
typedef __attribute__((ext_vector_type(8)))  float    v8f;
typedef __attribute__((ext_vector_type(4)))  float    v4f;

#define BB    4096
#define TT    512
#define D_IN  16
#define D_RNN 64
#define D_HID 128

union AU { v16h v; v8h h[2]; };

// Branch-free tanh: CDNA5 exposes v_tanh_f32 (confirmed lowering in round 2).
__device__ __forceinline__ float fast_tanh(float x) {
#if __has_builtin(__builtin_amdgcn_tanhf)
    return __builtin_amdgcn_tanhf(x);
#else
    float e = __builtin_amdgcn_exp2f(x * 2.885390081777927f); // exp(2x)
    return 1.0f - 2.0f * __builtin_amdgcn_rcpf(e + 1.0f);
#endif
}

// One fused RNN timestep.  LAST3 compiles in the hcat (MLP-input) stores for
// the final three timesteps only, keeping the 509-iteration hot loop branch-free.
template <bool LAST3>
__device__ __forceinline__ void rnn_step(
    int t, int lh, int ln, int cc, float biasc,
    v16h Bx, v16h Bh0, v16h Bh1,
    const float* __restrict__ xptr, v4f& xa, v4f& xb,
    _Float16 (*h_buf)[16 * D_RNN], float (*hcat)[3 * D_RNN])
{
    // A-layout for x_t: slots 0..7 = K(lh*8 + 0..7), slots 8..15 = K 16..31 (zero pad)
    v16h Ax = {};
    #pragma unroll
    for (int j = 0; j < 4; ++j) {
        Ax[j]     = (_Float16)xa[j];
        Ax[4 + j] = (_Float16)xb[j];
    }
    // software-pipeline next timestep's x load (always in range in the main loop)
    const int tn = LAST3 ? ((t + 1 < TT) ? t + 1 : TT - 1) : (t + 1);
    xa = *(const v4f*)(xptr + tn * D_IN);
    xb = *(const v4f*)(xptr + tn * D_IN + 4);

    v8f C = { biasc, biasc, biasc, biasc, biasc, biasc, biasc, biasc };
    // xw_t contribution (off the serial h-chain)
    C = __builtin_amdgcn_wmma_f32_16x16x32_f16(false, Ax, false, Bx,
                                               (short)0, C, false, false);
    // h_{t-1} contribution: A-layout via 4x ds_load_b128
    {
        const _Float16* hb = &h_buf[(t + 1) & 1][0] + ln * D_RNN + lh * 8;
        AU a0, a1;
        a0.h[0] = *(const v8h*)(hb);        // K = lh*8 + 0..7
        a0.h[1] = *(const v8h*)(hb + 16);   // K = 16 + lh*8 + 0..7
        a1.h[0] = *(const v8h*)(hb + 32);   // K = 32 + ...
        a1.h[1] = *(const v8h*)(hb + 48);   // K = 48 + ...
        C = __builtin_amdgcn_wmma_f32_16x16x32_f16(false, a0.v, false, Bh0,
                                                   (short)0, C, false, false);
        C = __builtin_amdgcn_wmma_f32_16x16x32_f16(false, a1.v, false, Bh1,
                                                   (short)0, C, false, false);
    }

    // h_t = tanh(C); write f16 tile to the other buffer.
    _Float16* hw = &h_buf[t & 1][0];
    #pragma unroll
    for (int r = 0; r < 8; ++r) {
        float hv = fast_tanh(C[r]);                    // C row m = r + 8*lh, col = cc
        hw[(r + 8 * lh) * D_RNN + cc] = (_Float16)hv;
        if constexpr (LAST3) {
            // MLP input = tanh(h) (reference re-applies tanh to the concatenation)
            hcat[r + 8 * lh][(t - (TT - 3)) * D_RNN + cc] = fast_tanh(hv);
        }
    }
    __syncthreads();   // one barrier/step; double buffering covers WAR+RAW
}

// One workgroup = 16 batch rows, 128 threads = 4 waves.
// Wave w computes hidden columns [16w, 16w+16) via v_wmma_f32_16x16x32_f16.
__global__ __launch_bounds__(128)
void rnn_fused(const float* __restrict__ x,  const float* __restrict__ Wx,
               const float* __restrict__ Wh, const float* __restrict__ b,
               const float* __restrict__ W1, const float* __restrict__ b1,
               const float* __restrict__ W2, const float* __restrict__ b2,
               float* __restrict__ out)
{
    __shared__ __align__(16) _Float16 h_buf[2][16 * D_RNN]; // double-buffered h_t (f16)
    __shared__ float hcat[16][3 * D_RNN];                   // tanh(h) for last 3 steps
    __shared__ float out2s[16][D_HID];                      // MLP hidden activations

    const int tid  = threadIdx.x;
    const int lane = tid & 31;
    const int wave = tid >> 5;      // 0..3 -> hidden-column tile
    const int lh   = lane >> 4;     // lane half (K interleave select)
    const int ln   = lane & 15;     // row m (A/C) or col n (B/C)
    const int r0   = blockIdx.x * 16;

    const int cc = wave * 16 + ln;  // this wave/lane's hidden column 0..63

    // h_{-1} = 0: zero the buffer read at t=0 (buf[1]); 1024 halfs, 8 per thread.
    {
        v8h z = {};
        *(v8h*)(&h_buf[1][tid * 8]) = z;
    }
    __syncthreads();

    // ---- Preload Wh in B-layout (32x16 per v16h): slots 0..7 hold K lh*8 + 0..7,
    //      slots 8..15 hold K 16 + lh*8 + 0..7.
    v16h Bh0, Bh1;
    #pragma unroll
    for (int s = 0; s < 16; ++s) {
        const int kk = (s < 8) ? (lh * 8 + s) : (16 + lh * 8 + (s - 8));
        Bh0[s] = (_Float16)Wh[kk * D_RNN + cc];        // K = 0..31
        Bh1[s] = (_Float16)Wh[(32 + kk) * D_RNN + cc];   // K = 32..63
    }
    // ---- Preload Wx in B-layout, K padded 16 -> 32 (upper half zero)
    v16h Bx = {};
    {
        #pragma unroll
        for (int s = 0; s < 8; ++s) Bx[s] = (_Float16)Wx[(lh * 8 + s) * D_RNN + cc];
    }
    const float biasc = b[cc];

    // ---- x pointer for this lane: row = r0+ln, K-half select = lh*8
    const float* xptr = x + (size_t)(r0 + ln) * (TT * D_IN) + lh * 8;
    v4f xa = *(const v4f*)(xptr);
    v4f xb = *(const v4f*)(xptr + 4);

    // Branch-free hot loop, then a 3-step tail that also records MLP inputs.
    for (int t = 0; t < TT - 3; ++t)
        rnn_step<false>(t, lh, ln, cc, biasc, Bx, Bh0, Bh1, xptr, xa, xb, h_buf, hcat);
    #pragma unroll
    for (int t = TT - 3; t < TT; ++t)
        rnn_step<true>(t, lh, ln, cc, biasc, Bx, Bh0, Bh1, xptr, xa, xb, h_buf, hcat);

    // ---- tiny MLP: [16 x 192] @ W1[192 x 128] + b1 -> tanh -> @ W2[128] + b2
    {
        const int j = tid;                 // hidden unit 0..127
        float accm[16];
        #pragma unroll
        for (int m = 0; m < 16; ++m) accm[m] = 0.f;
        for (int k = 0; k < 3 * D_RNN; ++k) {
            float w = W1[k * D_HID + j];
            #pragma unroll
            for (int m = 0; m < 16; ++m) accm[m] += hcat[m][k] * w;
        }
        float bb = b1[j];
        #pragma unroll
        for (int m = 0; m < 16; ++m) out2s[m][j] = fast_tanh(accm[m] + bb);
    }
    __syncthreads();

    if (tid < 16) {
        float s = b2[0];
        for (int jj = 0; jj < D_HID; ++jj) s += out2s[tid][jj] * W2[jj];
        out[r0 + tid] = s;
    }
}

extern "C" void kernel_launch(void* const* d_in, const int* in_sizes, int n_in,
                              void* d_out, int out_size, void* d_ws, size_t ws_size,
                              hipStream_t stream) {
    const float* x  = (const float*)d_in[0];
    const float* Wx = (const float*)d_in[1];
    const float* Wh = (const float*)d_in[2];
    const float* b  = (const float*)d_in[3];
    const float* W1 = (const float*)d_in[4];
    const float* b1 = (const float*)d_in[5];
    const float* W2 = (const float*)d_in[6];
    const float* b2 = (const float*)d_in[7];
    float* out = (float*)d_out;

    dim3 grid(BB / 16);   // 256 independent batch tiles hide the serial T chain
    dim3 block(128);      // 4 waves: one 16-col WMMA tile each
    hipLaunchKernelGGL(rnn_fused, grid, block, 0, stream,
                       x, Wx, Wh, b, W1, b1, W2, b2, out);
}
